// MultiHeadAttention_86217173500225
// MI455X (gfx1250) — matrix operand order repaired and checked by reading, not compiled, not run on hardware
//
#include <hip/hip_runtime.h>
#include <hip/hip_bf16.h>
#include <math.h>
#include <stdint.h>

typedef __attribute__((ext_vector_type(16))) __bf16 bf16x16;
typedef __attribute__((ext_vector_type(8)))  __bf16 bf16x8;
typedef __attribute__((ext_vector_type(8)))  float  f32x8;
typedef __attribute__((ext_vector_type(4)))  unsigned int v4u;
typedef __attribute__((ext_vector_type(4)))  int v4i;
typedef __attribute__((ext_vector_type(8)))  int v8i;

#define DEVINL __device__ __forceinline__

#if defined(__HIP_DEVICE_COMPILE__) && __has_builtin(__builtin_amdgcn_tensor_load_to_lds)
#define HAVE_TDM 1
#if __has_include(<hip/amd_detail/amd_gfx1250_TDM.h>)
#define TDM_6ARG 1
#else
#define TDM_6ARG 0
#endif
#else
#define HAVE_TDM 0
#endif

static constexpr int B_  = 8;
static constexpr int S_  = 2048;
static constexpr int E_  = 1024;
static constexpr int H_  = 8;
static constexpr int HS_ = 128;
static constexpr int FF_ = 4096;
static constexpr int M_  = B_ * S_;   // 16384 tokens

// ---------------------------------------------------------------- WMMA helpers

DEVINL f32x8 wmma_bf16(bf16x16 a, bf16x16 b, f32x8 c) {
  // D = A(16x32) * B(32x16) + C, f32 accumulate
  return __builtin_amdgcn_wmma_f32_16x16x32_bf16(false, a, false, b, (short)0, c,
                                                 false, false);
}

DEVINL f32x8 zero8() {
  f32x8 z;
#pragma unroll
  for (int i = 0; i < 8; ++i) z[i] = 0.0f;
  return z;
}

// A-fragment (16x32 bf16, row-major source, row stride ldm elements).
// lane<16:  m=lane,    K = {0..15}
// lane>=16: m=lane-16, K = {16..31}
DEVINL bf16x16 load_a_frag(const __bf16* base, int ldm, int lane) {
  const int m  = lane & 15;
  const int hh = lane >> 4;
  const __bf16* p = base + (size_t)m * ldm + hh * 16;
  bf16x8 lo = *(const bf16x8*)(p);
  bf16x8 hi = *(const bf16x8*)(p + 8);
  bf16x16 a;
#pragma unroll
  for (int i = 0; i < 8; ++i) { a[i] = lo[i]; a[i + 8] = hi[i]; }
  return a;
}

// B-fragment (32x16) where the K dimension is contiguous in memory per column n:
// element(k,n) at base + n*ldn + k.  lane<16: n=lane, k=0..15 ; lane>=16: n=lane-16, k=16..31
DEVINL bf16x16 load_b_kcontig(const __bf16* base, int ldn, int lane) {
  const __bf16* p = base + (size_t)(lane & 15) * ldn + (lane >> 4) * 16;
  return *(const bf16x16*)p;
}

// B-fragment from pre-packed weights: tile = 512 bf16, lane-major, 16 contiguous per lane.
DEVINL bf16x16 load_b_packed(const __bf16* packed, int tileIdx, int lane) {
  return *(const bf16x16*)(packed + ((size_t)tileIdx * 32 + lane) * 16);
}

// 16-lane row reductions (rows live on 16 lanes within each half-wave)
DEVINL float rmax16(float v) {
  v = fmaxf(v, __shfl_xor(v, 1, 32));
  v = fmaxf(v, __shfl_xor(v, 2, 32));
  v = fmaxf(v, __shfl_xor(v, 4, 32));
  v = fmaxf(v, __shfl_xor(v, 8, 32));
  return v;
}
DEVINL float rsum16(float v) {
  v += __shfl_xor(v, 1, 32);
  v += __shfl_xor(v, 2, 32);
  v += __shfl_xor(v, 4, 32);
  v += __shfl_xor(v, 8, 32);
  return v;
}

// ---------------------------------------------------------------- TDM helper
#if HAVE_TDM
// 1-D contiguous DMA of (n8 * 8) bytes from global to LDS via the Tensor Data
// Mover. D# per CDNA5 ISA ch.8: group0 = {count=1, lds_addr, global_addr,
// type=2}; group1 = {data_size=8B, tensor_dim0=tile_dim0=n8, dim1=1,
// stride0=n8}; groups 2/3 zero (<=2D tensor). Tracked with TENSORcnt.
DEVINL void tdm_load_1d(unsigned lds_off, const void* gptr, unsigned n8) {
  unsigned long long ga = (unsigned long long)(uintptr_t)gptr;
  v4u g0;
  g0.x = 0x1u;                                   // count=1, user mode
  g0.y = lds_off;                                // lds_addr
  g0.z = (unsigned)(ga & 0xFFFFFFFFu);           // global_addr[31:0]
  g0.w = (unsigned)((ga >> 32) & 0x01FFFFFFu)    // global_addr[56:32]
         | (2u << 30);                           // type = 2 ("image")
  v8i g1;
  g1[0] = (int)(3u << 16);                       // wg_mask=0, data_size=3 (8B)
  g1[1] = (int)((n8 & 0xFFFFu) << 16);           // tensor_dim0[15:0]
  g1[2] = (int)((n8 >> 16) | (1u << 16));        // tensor_dim0[31:16], dim1=1
  g1[3] = (int)((n8 & 0xFFFFu) << 16);           // tile_dim0 = n8
  g1[4] = 0;                                     // tile_dim1=0, tile_dim2=0
  g1[5] = (int)n8;                               // tensor_dim0_stride[31:0]
  g1[6] = 0;
  g1[7] = 0;
  v4i z4 = {0, 0, 0, 0};
  asm volatile("" ::: "memory");
#if TDM_6ARG
  v8i z8 = {0, 0, 0, 0, 0, 0, 0, 0};
  __builtin_amdgcn_tensor_load_to_lds(g0, g1, z4, z4, z8, 0);
#else
  __builtin_amdgcn_tensor_load_to_lds(g0, g1, z4, z4, 0);
#endif
}
#endif

// ---------------------------------------------------------------- prep kernels

__global__ void k_f32_to_bf16(const float* __restrict__ src,
                              __bf16* __restrict__ dst, int n) {
  int i = blockIdx.x * blockDim.x + threadIdx.x;
  if (i < n) dst[i] = (__bf16)src[i];
}

// Pack a (batched) row-major f32 [K,N] weight into WMMA B-fragment tiles (bf16).
__global__ void k_pack_b(const float* __restrict__ W, __bf16* __restrict__ dst,
                         int K, int N, int total) {
  int i = blockIdx.x * blockDim.x + threadIdx.x;
  if (i >= total) return;
  int kn   = K * N;
  int bidx = i / kn;
  int rem  = i - bidx * kn;
  int tile = rem >> 9;
  int lane = (rem >> 4) & 31;
  int j    = rem & 15;
  int nt   = tile % (N / 16);
  int tk   = tile / (N / 16);
  int n    = nt * 16 + (lane & 15);
  int k    = tk * 32 + (lane >> 4) * 16 + j;
  dst[i] = (__bf16)W[(size_t)bidx * kn + (size_t)k * N + n];
}

// ---------------------------------------------------------------- QKV kernel
// One wave: 16 tokens x one head. q,k -> [B,H,S,HS] row-major; v -> [B,H,HS,S].
__global__ void k_qkv(const __bf16* __restrict__ xb,
                      const __bf16* __restrict__ wqp, const __bf16* __restrict__ wkp,
                      const __bf16* __restrict__ wvp,
                      const float* __restrict__ bq, const float* __restrict__ bk,
                      const float* __restrict__ bv,
                      __bf16* __restrict__ qo, __bf16* __restrict__ ko,
                      __bf16* __restrict__ vto) {
  const int lane = threadIdx.x & 31;
  const int s0 = blockIdx.x * 16, h = blockIdx.y, b = blockIdx.z;
  const int bh = b * H_ + h;
  const int n = lane & 15, hh = lane >> 4;

  const __bf16* xbase = xb + ((size_t)(b * S_ + s0)) * E_ + h * HS_;
  bf16x16 xa[4];
#pragma unroll
  for (int kc = 0; kc < 4; ++kc) xa[kc] = load_a_frag(xbase + kc * 32, E_, lane);

#pragma unroll
  for (int w = 0; w < 2; ++w) {
    const __bf16* wp   = (w == 0 ? wqp : wkp) + (size_t)h * HS_ * HS_;
    const float*  bias = (w == 0 ? bq  : bk ) + h * HS_;
    __bf16*       out  = (w == 0 ? qo  : ko ) + (size_t)bh * S_ * HS_;
#pragma unroll
    for (int nt = 0; nt < 8; ++nt) {
      f32x8 c = zero8();
#pragma unroll
      for (int kc = 0; kc < 4; ++kc)
        c = wmma_bf16(xa[kc], load_b_packed(wp, kc * 8 + nt, lane), c);
      float bb = bias[nt * 16 + n];
#pragma unroll
      for (int r = 0; r < 8; ++r)
        out[(size_t)(s0 + r + 8 * hh) * HS_ + nt * 16 + n] = (__bf16)(c[r] + bb);
    }
  }

  {
    const __bf16* wp   = wvp + (size_t)h * HS_ * HS_;
    const float*  bias = bv + h * HS_;
    __bf16*       out  = vto + (size_t)bh * HS_ * S_;
#pragma unroll
    for (int nt = 0; nt < 8; ++nt) {
      f32x8 c = zero8();
#pragma unroll
      for (int kc = 0; kc < 4; ++kc)
        c = wmma_bf16(xa[kc], load_b_packed(wp, kc * 8 + nt, lane), c);
      float bb = bias[nt * 16 + n];
      bf16x8 v8;
#pragma unroll
      for (int r = 0; r < 8; ++r) v8[r] = (__bf16)(c[r] + bb);
      *(bf16x8*)(out + (size_t)(nt * 16 + n) * S_ + s0 + 8 * hh) = v8;
    }
  }
}

// ---------------------------------------------------------------- flash attention
__global__ void k_attn(const __bf16* __restrict__ Q, const __bf16* __restrict__ K,
                       const __bf16* __restrict__ VT, __bf16* __restrict__ O) {
  __shared__ __align__(32) __bf16 pscr[16 * 40];   // 16x32 P tile, padded rows
  const int lane = threadIdx.x & 31;
  const int s0 = blockIdx.x * 16, h = blockIdx.y, b = blockIdx.z;
  const int bh = b * H_ + h;
  const int n = lane & 15, hh = lane >> 4;

  const __bf16* q  = Q  + (size_t)bh * S_ * HS_;
  const __bf16* kk = K  + (size_t)bh * S_ * HS_;
  const __bf16* vt = VT + (size_t)bh * HS_ * S_;

  bf16x16 qa[4];
#pragma unroll
  for (int kc = 0; kc < 4; ++kc)
    qa[kc] = load_a_frag(q + (size_t)s0 * HS_ + kc * 32, HS_, lane);

  f32x8 acc[8];
#pragma unroll
  for (int t = 0; t < 8; ++t) acc[t] = zero8();
  float mr[8], lr[8];
#pragma unroll
  for (int r = 0; r < 8; ++r) { mr[r] = -__builtin_inff(); lr[r] = 0.0f; }

  const float scale = 0.08838834764831845f;   // 1/sqrt(128)

  for (int sk = 0; sk < S_; sk += 32) {
    __builtin_prefetch(kk + (size_t)(sk + 32) * HS_, 0, 0);
    f32x8 sc0 = zero8(), sc1 = zero8();
#pragma unroll
    for (int kc = 0; kc < 4; ++kc) {
      sc0 = wmma_bf16(qa[kc],
                      load_b_kcontig(kk + (size_t)sk * HS_ + kc * 32, HS_, lane), sc0);
      sc1 = wmma_bf16(qa[kc],
                      load_b_kcontig(kk + (size_t)(sk + 16) * HS_ + kc * 32, HS_, lane), sc1);
    }
    float alpha[8];
#pragma unroll
    for (int r = 0; r < 8; ++r) {
      float v0 = sc0[r] * scale, v1 = sc1[r] * scale;
      float mloc = rmax16(fmaxf(v0, v1));
      float mnew = fmaxf(mr[r], mloc);
      float a  = __expf(mr[r] - mnew);
      float p0 = __expf(v0 - mnew);
      float p1 = __expf(v1 - mnew);
      lr[r] = lr[r] * a + rsum16(p0 + p1);
      mr[r] = mnew;
      alpha[r] = a;
      pscr[(r + 8 * hh) * 40 + n]      = (__bf16)p0;
      pscr[(r + 8 * hh) * 40 + 16 + n] = (__bf16)p1;
    }
#pragma unroll
    for (int t = 0; t < 8; ++t)
#pragma unroll
      for (int r = 0; r < 8; ++r) acc[t][r] *= alpha[r];

    bf16x16 pa = load_a_frag(pscr, 40, lane);
#pragma unroll
    for (int dt = 0; dt < 8; ++dt)
      acc[dt] = wmma_bf16(pa,
                          load_b_kcontig(vt + (size_t)(dt * 16) * S_ + sk, S_, lane),
                          acc[dt]);
  }

  float rinv[8];
#pragma unroll
  for (int r = 0; r < 8; ++r) rinv[r] = 1.0f / lr[r];
#pragma unroll
  for (int dt = 0; dt < 8; ++dt)
#pragma unroll
    for (int r = 0; r < 8; ++r) {
      size_t idx = (size_t)(b * S_ + s0 + r + 8 * hh) * E_ + h * HS_ + dt * 16 + n;
      O[idx] = (__bf16)(acc[dt][r] * rinv[r]);
    }
}

// ---------------------------------------------------------------- MLP GEMM
// One wave: 16 x 64 tile. B tiles for one K-step are a contiguous 4KB block in
// the packed layout -> double-buffered TDM staging to LDS, WMMA feeds from LDS.
__global__ void k_mlp(const __bf16* __restrict__ A, const __bf16* __restrict__ Bp,
                      const float* __restrict__ bias, int Kdim, int Ndim,
                      __bf16* __restrict__ outb, float* __restrict__ outf, int relu) {
  const int lane = threadIdx.x & 31;
  const int row0 = blockIdx.x * 16;
  const int n0   = blockIdx.y * 64;
  const int ntiles = Ndim / 16;
  const int nk = Kdim >> 5;
  const __bf16* abase = A + (size_t)row0 * Kdim;

  f32x8 acc[4];
#pragma unroll
  for (int j = 0; j < 4; ++j) acc[j] = zero8();

#if HAVE_TDM
  __shared__ __align__(128) __bf16 bstage[2][2048];   // 2 x 4KB double buffer
  const unsigned lds0 = (unsigned)(uintptr_t)(void*)&bstage[0][0];
  const unsigned lds1 = (unsigned)(uintptr_t)(void*)&bstage[1][0];
  // prologue: stage K-step 0 (4 tiles = 512 x 8B)
  tdm_load_1d(lds0, Bp + ((size_t)0 * ntiles + (n0 >> 4)) * 512, 512);
#endif

  for (int t = 0; t < nk; ++t) {
    const int kc = t * 32;
    __builtin_prefetch(abase + kc + 128, 0, 0);
#if HAVE_TDM
    if (t + 1 < nk) {
      tdm_load_1d(((t + 1) & 1) ? lds1 : lds0,
                  Bp + ((size_t)(t + 1) * ntiles + (n0 >> 4)) * 512, 512);
      __builtin_amdgcn_s_wait_tensorcnt(1);
    } else {
      __builtin_amdgcn_s_wait_tensorcnt(0);
    }
    asm volatile("" ::: "memory");
    const __bf16* bsrc = (t & 1) ? &bstage[1][0] : &bstage[0][0];
#else
    const __bf16* bsrc = Bp + ((size_t)t * ntiles + (n0 >> 4)) * 512;
#endif
    bf16x16 a = load_a_frag(abase + kc, Kdim, lane);
#pragma unroll
    for (int j = 0; j < 4; ++j)
      acc[j] = wmma_bf16(a, load_b_packed(bsrc, j, lane), acc[j]);
  }

  const int n = lane & 15, hh = lane >> 4;
#pragma unroll
  for (int j = 0; j < 4; ++j) {
    int ng = n0 + j * 16 + n;
    float bb = bias[ng];
#pragma unroll
    for (int r = 0; r < 8; ++r) {
      float v = acc[j][r] + bb;
      if (relu) v = fmaxf(v, 0.0f);
      size_t idx = (size_t)(row0 + r + 8 * hh) * Ndim + ng;
      if (outb) outb[idx] = (__bf16)v;
      else      outf[idx] = v;
    }
  }
}

// ---------------------------------------------------------------- launch

extern "C" void kernel_launch(void* const* d_in, const int* in_sizes, int n_in,
                              void* d_out, int out_size, void* d_ws, size_t ws_size,
                              hipStream_t stream) {
  (void)in_sizes; (void)n_in; (void)out_size; (void)ws_size;

  const float* x  = (const float*)d_in[0];
  const float* Wq = (const float*)d_in[1];
  const float* bq = (const float*)d_in[2];
  const float* Wk = (const float*)d_in[3];
  const float* bk = (const float*)d_in[4];
  const float* Wv = (const float*)d_in[5];
  const float* bv = (const float*)d_in[6];
  const float* W1 = (const float*)d_in[7];
  const float* b1 = (const float*)d_in[8];
  const float* W2 = (const float*)d_in[9];
  const float* b2 = (const float*)d_in[10];

  const size_t XBe = (size_t)B_ * S_ * E_;        // 16,777,216 elems
  __bf16* xb  = (__bf16*)d_ws;                    // [B,S,E]
  __bf16* qb  = xb  + XBe;                        // [B,H,S,HS]
  __bf16* kb  = qb  + XBe;                        // [B,H,S,HS]
  __bf16* vtb = kb  + XBe;                        // [B,H,HS,S]
  __bf16* ob  = vtb + XBe;                        // [B*S, E] attention out
  __bf16* wqp = ob  + XBe;
  __bf16* wkp = wqp + (size_t)H_ * HS_ * HS_;
  __bf16* wvp = wkp + (size_t)H_ * HS_ * HS_;
  __bf16* w1p = wvp + (size_t)H_ * HS_ * HS_;
  __bf16* w2p = w1p + (size_t)E_ * FF_;
  __bf16* hb  = xb;   // hidden [M, FF] aliases x/q/k/vT region (exactly 4*XBe)

  const int nx = (int)XBe;
  k_f32_to_bf16<<<(nx + 255) / 256, 256, 0, stream>>>(x, xb, nx);

  const int nwh = H_ * HS_ * HS_;
  k_pack_b<<<(nwh + 255) / 256, 256, 0, stream>>>(Wq, wqp, HS_, HS_, nwh);
  k_pack_b<<<(nwh + 255) / 256, 256, 0, stream>>>(Wk, wkp, HS_, HS_, nwh);
  k_pack_b<<<(nwh + 255) / 256, 256, 0, stream>>>(Wv, wvp, HS_, HS_, nwh);
  const int nw1 = E_ * FF_;
  k_pack_b<<<(nw1 + 255) / 256, 256, 0, stream>>>(W1, w1p, E_, FF_, nw1);
  k_pack_b<<<(nw1 + 255) / 256, 256, 0, stream>>>(W2, w2p, FF_, E_, nw1);

  dim3 gAttn(S_ / 16, H_, B_);
  k_qkv <<<gAttn, 32, 0, stream>>>(xb, wqp, wkp, wvp, bq, bk, bv, qb, kb, vtb);
  k_attn<<<gAttn, 32, 0, stream>>>(qb, kb, vtb, ob);

  k_mlp<<<dim3(M_ / 16, FF_ / 64), 32, 0, stream>>>(ob, w1p, b1, E_, FF_, hb, nullptr, 1);
  k_mlp<<<dim3(M_ / 16, E_ / 64),  32, 0, stream>>>(hb, w2p, b2, FF_, E_, nullptr,
                                                    (float*)d_out, 0);
}
